// GCN_unsupervised_48129403519138
// MI455X (gfx1250) — compile-verified, matrix operand order checked by reading, not run on hardware
//
#include <hip/hip_runtime.h>
#include <hip/hip_bf16.h>

#define HID 128
#define TILE_ROWS 128

typedef __attribute__((ext_vector_type(16))) _Float16 v16h;
typedef __attribute__((ext_vector_type(8)))  float    v8f;
typedef __attribute__((ext_vector_type(4)))  unsigned int v4u;
typedef __attribute__((ext_vector_type(8)))  int      v8i;
typedef __attribute__((ext_vector_type(4)))  int      v4i;

#if __has_builtin(__builtin_amdgcn_tensor_load_to_lds) && __has_builtin(__builtin_amdgcn_s_wait_tensorcnt)
#define GCN_HAS_TDM 1
#else
#define GCN_HAS_TDM 0
#endif

// ---------------------------------------------------------------------------
// Dense GEMM  C[M,128] = A[M,128] @ W[128,128]  via v_wmma_f32_16x16x32_f16.
// Block = 256 threads = 8 waves; block computes 128(M) x 128(N).
// Wave w owns rows [16w,16w+16), loops over 8 N-tiles; A fragments built once
// per wave and reused.  A tile staged to LDS as f32 by the Tensor Data Mover
// (hardware zero-fill for the M tail via tensor_dim1); W staged transposed
// as f16 by VALU (amortized over 128 rows).
// ---------------------------------------------------------------------------
__global__ __launch_bounds__(256) void gcn_gemm_wmma(
    const float* __restrict__ A, const float* __restrict__ W,
    float* __restrict__ C, int M)
{
    __shared__ float    sAf[TILE_ROWS * HID];   // 64 KB, row-major f32 (TDM dest)
    __shared__ _Float16 sWt[HID * HID];         // 32 KB, W transposed f16

    const int t = threadIdx.x;
    const int rowBase = blockIdx.x * TILE_ROWS;

    // ---- Stage A tile via Tensor Data Mover (wave 0 issues; EXEC ignored) ----
#if GCN_HAS_TDM
    if (t < 32) {
        const unsigned long long ga =
            (unsigned long long)(uintptr_t)A + (unsigned long long)rowBase * (HID * 4);
        // LDS aperture: low 32 bits of the flat address == LDS byte offset.
        const unsigned ldsA = (unsigned)(uintptr_t)&sAf[0];
        const int rowsLeft = M - rowBase;          // OOB rows zero-filled by TDM
        v4u g0; v8i g1; v4i g2 = {}; v4i g3 = {};
        g0[0] = 1u;                                            // count=1 (valid user D#)
        g0[1] = ldsA;                                          // lds_addr
        g0[2] = (unsigned)(ga & 0xFFFFFFFFull);                // global_addr[31:0]
        g0[3] = (unsigned)((ga >> 32) & 0x01FFFFFFull)         // global_addr[56:32]
              | (2u << 30);                                    // type=2 ("image")
        g1[0] = (2 << 16);                                     // data_size=2 -> 4B
        g1[1] = (HID & 0xFFFF) << 16;                          // tensor_dim0 lo16 = 128
        g1[2] = (rowsLeft & 0xFFFF) << 16;                     // dim0 hi16=0 | dim1 lo16
        g1[3] = ((rowsLeft >> 16) & 0xFFFF) | (HID << 16);     // dim1 hi16 | tile_dim0=128
        g1[4] = TILE_ROWS;                                     // tile_dim1=128, tile_dim2=0
        g1[5] = HID;                                           // tensor_dim0_stride lo32
        g1[6] = 0;                                             // stride hi | dim1_stride lo
        g1[7] = 0;                                             // dim1_stride hi (unused, 2D)
#if __clang_major__ >= 23
        v8i g4 = {};
        __builtin_amdgcn_tensor_load_to_lds(g0, g1, g2, g3, g4, 0);
#else
        __builtin_amdgcn_tensor_load_to_lds(g0, g1, g2, g3, 0);
#endif
        __builtin_amdgcn_s_wait_tensorcnt(0);
    }
#else
    for (int i = t; i < TILE_ROWS * HID; i += 256) {
        const int row = rowBase + (i >> 7);
        sAf[i] = (row < M) ? A[(size_t)row * HID + (i & (HID - 1))] : 0.0f;
    }
#endif

    // ---- Stage W transposed into LDS, f32 -> f16 (LDS-contiguous writes) ----
    for (int i = t; i < HID * HID; i += 256) {
        const int n = i >> 7;
        const int k = i & (HID - 1);
        sWt[i] = (_Float16)W[(size_t)k * HID + n];
    }
    __syncthreads();

    const int wave = t >> 5;
    const int lane = t & 31;
    const int nlo  = lane & 15;    // M index (A frag) / N index (B,C frags)
    const int hi   = lane >> 4;    // lane-half: K offset +8 / M offset +8
    const int arow = 16 * wave + nlo;     // A-fragment row within block tile
    const int wrow = rowBase + 16 * wave; // first output row of this wave
    const bool fullTile = (wrow + 16 <= M);

    // Build the 4 A fragments (K chunks of 32) once; reuse across N-tiles.
    v16h afrag[4];
#pragma unroll
    for (int kc = 0; kc < HID / 32; ++kc) {
#pragma unroll
        for (int j = 0; j < 8; ++j) {
            const int kk = kc * 32 + ((j & 4) ? 16 : 0) + ((j & 3) * 2) + hi * 8;
            afrag[kc][2 * j]     = (_Float16)sAf[arow * HID + kk];
            afrag[kc][2 * j + 1] = (_Float16)sAf[arow * HID + kk + 1];
        }
    }

    for (int nt = 0; nt < HID / 16; ++nt) {
        v8f acc = {};
#pragma unroll
        for (int kc = 0; kc < HID / 32; ++kc) {
            v16h b;
#pragma unroll
            for (int j = 0; j < 8; ++j) {
                const int kk = kc * 32 + ((j & 4) ? 16 : 0) + ((j & 3) * 2) + hi * 8;
                b[2 * j]     = sWt[(nt * 16 + nlo) * HID + kk];
                b[2 * j + 1] = sWt[(nt * 16 + nlo) * HID + kk + 1];
            }
            acc = __builtin_amdgcn_wmma_f32_16x16x32_f16(
                      false, afrag[kc], false, b, (short)0, acc, false, false);
        }
        // C/D layout: VGPR r -> row (r + 8*hi), col (nt*16 + nlo)
        float* cb = C + (size_t)(wrow + 8 * hi) * HID + nt * 16 + nlo;
        if (fullTile) {
#pragma unroll
            for (int r = 0; r < 8; ++r) cb[(size_t)r * HID] = acc[r];
        } else {
#pragma unroll
            for (int r = 0; r < 8; ++r)
                if (wrow + 8 * hi + r < M) cb[(size_t)r * HID] = acc[r];
        }
    }
}

// ---------------------------------------------------------------------------
// Degree accumulation (float atomics; self-loop folded into rsqrt as +1).
// ---------------------------------------------------------------------------
__global__ __launch_bounds__(256) void gcn_degree(
    const int* __restrict__ dst, float* __restrict__ deg, int E)
{
    const int e = blockIdx.x * blockDim.x + threadIdx.x;
    if (e < E) atomicAdd(&deg[dst[e]], 1.0f);
}

__global__ __launch_bounds__(256) void gcn_rsqrt(float* __restrict__ deg, int N)
{
    const int i = blockIdx.x * blockDim.x + threadIdx.x;
    if (i < N) deg[i] = rsqrtf(deg[i] + 1.0f);   // +1 = self loop
}

// ---------------------------------------------------------------------------
// Edge aggregation: out[dst] += xw[src] * dinv[src]*dinv[dst].
// One wave per edge (edge id is wave-uniform -> scalar index/dinv loads);
// each lane handles a float4 (32 lanes * 4 = 128 feats).
// Edges e in [E, E+N) are the implicit self-loops.  XW/out are L2-resident.
// ---------------------------------------------------------------------------
__global__ __launch_bounds__(256) void gcn_scatter(
    const float* __restrict__ xw, const float* __restrict__ dinv,
    const int* __restrict__ src, const int* __restrict__ dst,
    float* __restrict__ out, int E, int N)
{
    const long long e = (long long)blockIdx.x * 8 + (threadIdx.x >> 5); // uniform
    const int lane = threadIdx.x & 31;
    if (e >= (long long)E + N) return;

    int s, d;
    if (e < E) { s = src[e]; d = dst[e]; }
    else       { s = d = (int)(e - (long long)E); }

    const float norm = dinv[s] * dinv[d];
    const float4 v = *(const float4*)(xw + (size_t)s * HID + lane * 4);
    float* o = out + (size_t)d * HID + lane * 4;
    atomicAdd(o + 0, v.x * norm);
    atomicAdd(o + 1, v.y * norm);
    atomicAdd(o + 2, v.z * norm);
    atomicAdd(o + 3, v.w * norm);
}

__global__ __launch_bounds__(256) void gcn_bias_relu(
    float* __restrict__ h, const float* __restrict__ b, long long NH)
{
    const long long i = (long long)blockIdx.x * blockDim.x + threadIdx.x;
    if (i < NH) {
        const float v = h[i] + b[i & (HID - 1)];
        h[i] = v > 0.0f ? v : 0.0f;
    }
}

// ---------------------------------------------------------------------------
// global_mean_pool: scatter-add features + counts over dict_node segments.
// ---------------------------------------------------------------------------
__global__ __launch_bounds__(256) void gcn_pool_scatter(
    const float* __restrict__ h, const int* __restrict__ seg,
    float* __restrict__ sums, float* __restrict__ cnt, int N)
{
    const long long i = (long long)blockIdx.x * 8 + (threadIdx.x >> 5); // uniform
    const int lane = threadIdx.x & 31;
    if (i >= N) return;
    const int g = seg[i];
    const float4 v = *(const float4*)(h + (size_t)i * HID + lane * 4);
    float* o = sums + (size_t)g * HID + lane * 4;
    atomicAdd(o + 0, v.x);
    atomicAdd(o + 1, v.y);
    atomicAdd(o + 2, v.z);
    atomicAdd(o + 3, v.w);
    if (lane == 0) atomicAdd(&cnt[g], 1.0f);
}

__global__ __launch_bounds__(256) void gcn_pool_div(
    float* __restrict__ z, const float* __restrict__ cnt, long long NH)
{
    const long long i = (long long)blockIdx.x * blockDim.x + threadIdx.x;
    if (i < NH) {
        const float c = cnt[i >> 7];
        z[i] = z[i] / fmaxf(c, 1.0f);
    }
}

// ---------------------------------------------------------------------------
// Decode: per-edge dot product over 128 features; wave-per-edge with a
// wave32 shuffle-xor reduction.  z is L2-resident.
// ---------------------------------------------------------------------------
__global__ __launch_bounds__(256) void gcn_decode(
    const float* __restrict__ z, const int* __restrict__ pe,
    const int* __restrict__ ne, float* __restrict__ out, int EP)
{
    const long long e = (long long)blockIdx.x * 8 + (threadIdx.x >> 5); // uniform
    const int lane = threadIdx.x & 31;
    if (e >= 2LL * EP) return;

    int a, b;
    if (e < EP) { a = pe[e];                   b = pe[EP + (int)e]; }
    else        { const int o = (int)(e - EP); a = ne[o]; b = ne[EP + o]; }

    const float4 va = *(const float4*)(z + (size_t)a * HID + lane * 4);
    const float4 vb = *(const float4*)(z + (size_t)b * HID + lane * 4);
    float p = va.x * vb.x + va.y * vb.y + va.z * vb.z + va.w * vb.w;

#pragma unroll
    for (int off = 16; off > 0; off >>= 1)
        p += __shfl_xor(p, off, 32);

    if (lane == 0) out[e] = p;
}

// ---------------------------------------------------------------------------
extern "C" void kernel_launch(void* const* d_in, const int* in_sizes, int n_in,
                              void* d_out, int out_size, void* d_ws, size_t ws_size,
                              hipStream_t stream) {
    (void)n_in; (void)out_size; (void)ws_size;

    const float* x   = (const float*)d_in[0];
    const float* W1  = (const float*)d_in[1];
    const float* b1  = (const float*)d_in[2];
    const float* W2  = (const float*)d_in[3];
    const float* b2  = (const float*)d_in[4];
    const int* ei    = (const int*)d_in[5];   // [2,E]: src then dst
    const int* dict  = (const int*)d_in[6];
    const int* pe    = (const int*)d_in[7];   // [2,EP]
    const int* ne    = (const int*)d_in[8];   // [2,EP]

    const int N  = in_sizes[0] / HID;
    const int E  = in_sizes[5] / 2;
    const int EP = in_sizes[7] / 2;
    float* out = (float*)d_out;

    // Workspace carve (256B aligned slots)
    const size_t NA = ((size_t)N + 63) & ~(size_t)63;
    const size_t NH = (size_t)N * HID;
    float* ws   = (float*)d_ws;
    float* dinv = ws;                 // [N]   degree -> dinv (in place)
    float* cnt  = ws + NA;            // [N]   pool counts
    float* bufA = ws + 2 * NA;        // [N,H] GEMM output (messages source)
    float* bufB = bufA + NH;          // [N,H] aggregated features
    float* zbuf = bufB + NH;          // [N,H] pooled embeddings

    const unsigned B256 = 256u;
    const unsigned gE   = (unsigned)(((long long)E + 255) / 256);
    const unsigned gN   = (unsigned)((N + 255) / 256);
    const unsigned gNH  = (unsigned)(((long long)NH + 255) / 256);
    const unsigned gTile = (unsigned)((N + TILE_ROWS - 1) / TILE_ROWS);
    const unsigned gEdgeWave = (unsigned)(((long long)E + N + 7) / 8);
    const unsigned gNodeWave = (unsigned)(((long long)N + 7) / 8);
    const unsigned gDecWave  = (unsigned)((2LL * EP + 7) / 8);

    // ---- degree / symmetric norm ----
    hipMemsetAsync(dinv, 0, (size_t)N * sizeof(float), stream);
    gcn_degree<<<gE, B256, 0, stream>>>(ei + E, dinv, E);
    gcn_rsqrt<<<gN, B256, 0, stream>>>(dinv, N);

    // ---- layer 1: XW1 -> aggregate -> +b1, relu ----
    gcn_gemm_wmma<<<gTile, B256, 0, stream>>>(x, W1, bufA, N);
    hipMemsetAsync(bufB, 0, NH * sizeof(float), stream);
    gcn_scatter<<<gEdgeWave, B256, 0, stream>>>(bufA, dinv, ei, ei + E, bufB, E, N);
    gcn_bias_relu<<<gNH, B256, 0, stream>>>(bufB, b1, (long long)NH);

    // ---- layer 2: HW2 -> aggregate -> +b2, relu ----
    gcn_gemm_wmma<<<gTile, B256, 0, stream>>>(bufB, W2, bufA, N);
    hipMemsetAsync(bufB, 0, NH * sizeof(float), stream);
    gcn_scatter<<<gEdgeWave, B256, 0, stream>>>(bufA, dinv, ei, ei + E, bufB, E, N);
    gcn_bias_relu<<<gNH, B256, 0, stream>>>(bufB, b2, (long long)NH);

    // ---- global mean pool over dict_node ----
    hipMemsetAsync(zbuf, 0, NH * sizeof(float), stream);
    hipMemsetAsync(cnt, 0, (size_t)N * sizeof(float), stream);
    gcn_pool_scatter<<<gNodeWave, B256, 0, stream>>>(bufB, dict, zbuf, cnt, N);
    gcn_pool_div<<<gNH, B256, 0, stream>>>(zbuf, cnt, (long long)NH);

    // ---- decode: per-edge dot products ----
    gcn_decode<<<gDecWave, B256, 0, stream>>>(zbuf, pe, ne, out, EP);
}
